// GIN_7516192768968
// MI455X (gfx1250) — matrix operand order repaired and checked by reading, not compiled, not run on hardware
//
#include <hip/hip_runtime.h>
#include <hip/hip_bf16.h>
#include <stdint.h>

typedef __attribute__((ext_vector_type(16))) _Float16 v16h;
typedef __attribute__((ext_vector_type(8)))  float    v8f;

union HFrag { v16h h; uint32_t u[8]; };

#define HDIM 128
#define FDIM 74
#define KPAD 96
#define BN_EPS 1e-5f

// ---------------------------------------------------------------------------
// prep: f32->f16 weight convert+transpose (col-major so B-frags are contiguous),
// fold bias+BN into scale/shift, init graph accumulators with lb[1..4].
// ---------------------------------------------------------------------------
__global__ __launch_bounds__(256) void prep_kernel(
    const float* __restrict__ fW1, const float* __restrict__ fW2,
    const float* __restrict__ cW1, const float* __restrict__ cW2,
    const float* __restrict__ fb1, const float* __restrict__ fg1, const float* __restrict__ fbe1,
    const float* __restrict__ fm1, const float* __restrict__ fv1,
    const float* __restrict__ fb2, const float* __restrict__ fg2, const float* __restrict__ fbe2,
    const float* __restrict__ fm2, const float* __restrict__ fv2,
    const float* __restrict__ cb1, const float* __restrict__ cg1, const float* __restrict__ cbe1,
    const float* __restrict__ cm1, const float* __restrict__ cv1,
    const float* __restrict__ cb2, const float* __restrict__ cg2, const float* __restrict__ cbe2,
    const float* __restrict__ cm2, const float* __restrict__ cv2,
    const float* __restrict__ lb,
    _Float16* __restrict__ w1t, _Float16* __restrict__ w2t,
    _Float16* __restrict__ cw1t, _Float16* __restrict__ cw2t,
    float* __restrict__ ss, float* __restrict__ outAcc, int G)
{
    int id = blockIdx.x * 256 + threadIdx.x;
    // S0: first-layer W1, transposed [n][k], K padded 74->96 with zeros
    if (id < KPAD * HDIM) {
        int n = id / KPAD, k = id - n * KPAD;
        w1t[n * KPAD + k] = (_Float16)(k < FDIM ? fW1[k * HDIM + n] : 0.0f);
        return;
    }
    id -= KPAD * HDIM;
    if (id < HDIM * HDIM) {
        int n = id / HDIM, k = id - n * HDIM;
        w2t[n * HDIM + k] = (_Float16)fW2[k * HDIM + n];
        return;
    }
    id -= HDIM * HDIM;
    if (id < 4 * HDIM * HDIM) {
        int l = id / (HDIM * HDIM), r = id - l * (HDIM * HDIM);
        int n = r / HDIM, k = r - n * HDIM;
        cw1t[l * HDIM * HDIM + n * HDIM + k] = (_Float16)cW1[l * HDIM * HDIM + k * HDIM + n];
        return;
    }
    id -= 4 * HDIM * HDIM;
    if (id < 4 * HDIM * HDIM) {
        int l = id / (HDIM * HDIM), r = id - l * (HDIM * HDIM);
        int n = r / HDIM, k = r - n * HDIM;
        cw2t[l * HDIM * HDIM + n * HDIM + k] = (_Float16)cW2[l * HDIM * HDIM + k * HDIM + n];
        return;
    }
    id -= 4 * HDIM * HDIM;
    if (id < 10 * HDIM) {
        int set = id / HDIM, col = id - set * HDIM;
        const float *b, *g, *be, *m, *v;
        if (set == 0)      { b = fb1; g = fg1; be = fbe1; m = fm1; v = fv1; }
        else if (set == 1) { b = fb2; g = fg2; be = fbe2; m = fm2; v = fv2; }
        else {
            int l = (set - 2) >> 1, which = (set - 2) & 1;
            if (which == 0) { b = cb1 + l * HDIM; g = cg1 + l * HDIM; be = cbe1 + l * HDIM; m = cm1 + l * HDIM; v = cv1 + l * HDIM; }
            else            { b = cb2 + l * HDIM; g = cg2 + l * HDIM; be = cbe2 + l * HDIM; m = cm2 + l * HDIM; v = cv2 + l * HDIM; }
        }
        float sc = g[col] * rsqrtf(v[col] + BN_EPS);
        ss[set * 256 + col]       = sc;
        ss[set * 256 + 128 + col] = (b[col] - m[col]) * sc + be[col];
        return;
    }
    id -= 10 * HDIM;
    if (id < G) outAcc[id] = lb[1] + lb[2] + lb[3] + lb[4];
}

// ---------------------------------------------------------------------------
// Fused 2-layer MLP via v_wmma_f32_16x16x32_f16.
// Block = 256 thr (8 waves) handles 128 node rows; wave w -> rows 16w..16w+15,
// all 128 output columns as 8 accumulator tiles of 16x16.
// A-frag (16-bit A 16x32): lane m = lane&15, hi = lane>>4,
//   VGPR i holds K pair at kb = 2*i + (hi<<4).
// B-frag (mirror, lane = N): VGPR i holds K pair at kb = 2*i + (hi<<4);
//   weights stored col-major in LDS so each pair is one contiguous dword.
// ---------------------------------------------------------------------------
template<bool FIRST>
__global__ __launch_bounds__(256) void mlp_kernel(
    const float* __restrict__ in, const float* __restrict__ aggr,
    const float* __restrict__ epsArr, int layer,
    const _Float16* __restrict__ gw1t, const _Float16* __restrict__ gw2t,
    const float* __restrict__ ss1, const float* __restrict__ ss2,
    float* __restrict__ hout, int nNodes)
{
    constexpr int KIN = FIRST ? KPAD : HDIM;
    __shared__ __align__(16) _Float16 Abuf[HDIM * HDIM];   // stage-1 A, reused as stage-2 A
    __shared__ __align__(16) _Float16 W1t[HDIM * KIN];
    __shared__ __align__(16) _Float16 W2t[HDIM * HDIM];

    const int tid = threadIdx.x;
    const int node0 = blockIdx.x * 128;

    // stage weights (dword copies from pre-transposed f16 in ws)
    {
        const uint32_t* s1 = (const uint32_t*)gw1t;
        uint32_t* d1 = (uint32_t*)W1t;
        for (int i = tid; i < HDIM * KIN / 2; i += 256) d1[i] = s1[i];
        const uint32_t* s2 = (const uint32_t*)gw2t;
        uint32_t* d2 = (uint32_t*)W2t;
        for (int i = tid; i < HDIM * HDIM / 2; i += 256) d2[i] = s2[i];
    }
    // stage A (f32 -> f16), zero-pad K and rows past N
    if (FIRST) {
        for (int i = tid; i < 128 * KPAD; i += 256) {
            int r = i / KPAD, k = i - r * KPAD;
            int node = node0 + r;
            float v = (k < FDIM && node < nNodes) ? in[node * FDIM + k] : 0.0f;
            Abuf[r * KPAD + k] = (_Float16)v;
        }
    } else {
        float e1 = 1.0f + epsArr[layer];
        for (int i = tid; i < 128 * HDIM; i += 256) {
            int node = node0 + (i >> 7);
            int k = i & 127;
            float v = (node < nNodes) ? (e1 * in[node * HDIM + k] + aggr[node * HDIM + k]) : 0.0f;
            Abuf[i] = (_Float16)v;
        }
    }
    __syncthreads();

    const int wave = tid >> 5, lane = tid & 31;
    const int m = lane & 15, hi = lane >> 4;
    const int r0 = wave << 4;

    v8f acc[8] = {};

    // ---- GEMM1: [16 x KIN] x [KIN x 128] ----
    #pragma unroll
    for (int kt = 0; kt < KIN / 32; ++kt) {
        const int k0 = kt * 32;
        HFrag fa;
        const _Float16* arow = Abuf + (r0 + m) * KIN + k0 + (hi << 4);
        #pragma unroll
        for (int i = 0; i < 8; ++i)
            fa.u[i] = *(const uint32_t*)(arow + 2 * i);
        #pragma unroll
        for (int c = 0; c < 8; ++c) {
            HFrag fb;
            const _Float16* brow = W1t + (c * 16 + m) * KIN + k0 + (hi << 4);
            #pragma unroll
            for (int i = 0; i < 8; ++i)
                fb.u[i] = *(const uint32_t*)(brow + 2 * i);
            acc[c] = __builtin_amdgcn_wmma_f32_16x16x32_f16(
                false, fa.h, false, fb.h, (short)0, acc[c], false, false);
        }
    }
    __syncthreads();   // all waves finished reading stage-1 Abuf

    // fused bias+BN+ReLU -> f16 intermediate back into Abuf [row][col]
    #pragma unroll
    for (int c = 0; c < 8; ++c) {
        int col = c * 16 + m;
        float sc = ss1[col], sh = ss1[128 + col];
        #pragma unroll
        for (int i = 0; i < 8; ++i) {
            float y = fmaxf(acc[c][i] * sc + sh, 0.0f);
            int row = r0 + i + (hi << 3);
            Abuf[row * HDIM + col] = (_Float16)y;
        }
    }
    __syncthreads();

    // ---- GEMM2: [16 x 128] x [128 x 128] ----
    #pragma unroll
    for (int c = 0; c < 8; ++c) acc[c] = v8f{};
    #pragma unroll
    for (int kt = 0; kt < HDIM / 32; ++kt) {
        const int k0 = kt * 32;
        HFrag fa;
        const _Float16* arow = Abuf + (r0 + m) * HDIM + k0 + (hi << 4);
        #pragma unroll
        for (int i = 0; i < 8; ++i)
            fa.u[i] = *(const uint32_t*)(arow + 2 * i);
        #pragma unroll
        for (int c = 0; c < 8; ++c) {
            HFrag fb;
            const _Float16* brow = W2t + (c * 16 + m) * HDIM + k0 + (hi << 4);
            #pragma unroll
            for (int i = 0; i < 8; ++i)
                fb.u[i] = *(const uint32_t*)(brow + 2 * i);
            acc[c] = __builtin_amdgcn_wmma_f32_16x16x32_f16(
                false, fa.h, false, fb.h, (short)0, acc[c], false, false);
        }
    }

    // fused bias+BN+ReLU -> f32 h out
    #pragma unroll
    for (int c = 0; c < 8; ++c) {
        int col = c * 16 + m;
        float sc = ss2[col], sh = ss2[128 + col];
        #pragma unroll
        for (int i = 0; i < 8; ++i) {
            float y = fmaxf(acc[c][i] * sc + sh, 0.0f);
            int node = node0 + r0 + i + (hi << 3);
            if (node < nNodes) hout[node * HDIM + col] = y;
        }
    }
}

// ---------------------------------------------------------------------------
// Edge scatter-add: wave per edge; h fits in 192MB L2, atomics resolve in L2.
// ---------------------------------------------------------------------------
__global__ __launch_bounds__(256) void aggregate_kernel(
    const float* __restrict__ h, const int* __restrict__ ei,
    float* __restrict__ aggr, int E)
{
    int e = blockIdx.x * 8 + (threadIdx.x >> 5);
    if (e >= E) return;
    int lane = threadIdx.x & 31;
    int src = ei[e];
    int dst = ei[E + e];
    const float4 v = *(const float4*)(h + (size_t)src * HDIM + lane * 4);
    float* d = aggr + (size_t)dst * HDIM + lane * 4;
    __hip_atomic_fetch_add(d + 0, v.x, __ATOMIC_RELAXED, __HIP_MEMORY_SCOPE_AGENT);
    __hip_atomic_fetch_add(d + 1, v.y, __ATOMIC_RELAXED, __HIP_MEMORY_SCOPE_AGENT);
    __hip_atomic_fetch_add(d + 2, v.z, __ATOMIC_RELAXED, __HIP_MEMORY_SCOPE_AGENT);
    __hip_atomic_fetch_add(d + 3, v.w, __ATOMIC_RELAXED, __HIP_MEMORY_SCOPE_AGENT);
}

// ---------------------------------------------------------------------------
// Per-layer readout: wave per node, dot(h_node, lW[l]) (+lb for layer 0),
// wave32 shuffle reduce, one atomic per node into outAcc[batch[node]].
// ---------------------------------------------------------------------------
__global__ __launch_bounds__(256) void readout_kernel(
    const float* __restrict__ h, const int* __restrict__ batch,
    const float* __restrict__ lW, const float* __restrict__ lb,
    int lidx, int addLb, float* __restrict__ outAcc, int nNodes)
{
    int node = blockIdx.x * 8 + (threadIdx.x >> 5);
    if (node >= nNodes) return;
    int lane = threadIdx.x & 31;
    const float4 hv = *(const float4*)(h + (size_t)node * HDIM + lane * 4);
    const float4 wv = *(const float4*)(lW + lidx * HDIM + lane * 4);
    float v = hv.x * wv.x + hv.y * wv.y + hv.z * wv.z + hv.w * wv.w;
    #pragma unroll
    for (int off = 16; off > 0; off >>= 1) v += __shfl_xor(v, off, 32);
    if (lane == 0) {
        if (addLb) v += lb[lidx];
        __hip_atomic_fetch_add(&outAcc[batch[node]], v, __ATOMIC_RELAXED, __HIP_MEMORY_SCOPE_AGENT);
    }
}

__global__ __launch_bounds__(256) void sigmoid_kernel(
    const float* __restrict__ acc, float* __restrict__ out, int n)
{
    int i = blockIdx.x * 256 + threadIdx.x;
    if (i < n) out[i] = 1.0f / (1.0f + __expf(-acc[i]));
}

// ---------------------------------------------------------------------------
extern "C" void kernel_launch(void* const* d_in, const int* in_sizes, int n_in,
                              void* d_out, int out_size, void* d_ws, size_t ws_size,
                              hipStream_t stream)
{
    const float* x    = (const float*)d_in[0];
    const int*   ei   = (const int*)d_in[1];
    const int*   batch= (const int*)d_in[2];
    const float* fW1  = (const float*)d_in[3];
    const float* fb1  = (const float*)d_in[4];
    const float* fg1  = (const float*)d_in[5];
    const float* fbe1 = (const float*)d_in[6];
    const float* fm1  = (const float*)d_in[7];
    const float* fv1  = (const float*)d_in[8];
    const float* fW2  = (const float*)d_in[9];
    const float* fb2  = (const float*)d_in[10];
    const float* fg2  = (const float*)d_in[11];
    const float* fbe2 = (const float*)d_in[12];
    const float* fm2  = (const float*)d_in[13];
    const float* fv2  = (const float*)d_in[14];
    const float* eps  = (const float*)d_in[15];
    const float* cW1  = (const float*)d_in[16];
    const float* cb1  = (const float*)d_in[17];
    const float* cg1  = (const float*)d_in[18];
    const float* cbe1 = (const float*)d_in[19];
    const float* cm1  = (const float*)d_in[20];
    const float* cv1  = (const float*)d_in[21];
    const float* cW2  = (const float*)d_in[22];
    const float* cb2  = (const float*)d_in[23];
    const float* cg2  = (const float*)d_in[24];
    const float* cbe2 = (const float*)d_in[25];
    const float* cm2  = (const float*)d_in[26];
    const float* cv2  = (const float*)d_in[27];
    const float* lW   = (const float*)d_in[28];
    const float* lb   = (const float*)d_in[29];

    const int N = in_sizes[2];          // nodes (batch array length)
    const int E = in_sizes[1] / 2;      // edges
    const int G = out_size;             // graphs (T==1)

    // workspace carve (256B aligned)
    char* ws = (char*)d_ws;
    size_t off = 0;
    auto carve = [&](size_t bytes) {
        size_t o = off;
        off = (off + bytes + 255) & ~(size_t)255;
        return o;
    };
    float*    hA     = (float*)(ws + carve((size_t)N * HDIM * 4));
    float*    hB     = (float*)(ws + carve((size_t)N * HDIM * 4));
    float*    aggr   = (float*)(ws + carve((size_t)N * HDIM * 4));
    _Float16* w1t    = (_Float16*)(ws + carve(KPAD * HDIM * 2));
    _Float16* w2t    = (_Float16*)(ws + carve(HDIM * HDIM * 2));
    _Float16* cw1t   = (_Float16*)(ws + carve(4 * HDIM * HDIM * 2));
    _Float16* cw2t   = (_Float16*)(ws + carve(4 * HDIM * HDIM * 2));
    float*    ss     = (float*)(ws + carve(10 * 256 * 4));
    float*    outAcc = (float*)(ws + carve((size_t)G * 4));

    // prep: weights + BN folds + outAcc init
    {
        int total = KPAD * HDIM + HDIM * HDIM + 8 * HDIM * HDIM + 10 * HDIM + G;
        prep_kernel<<<(total + 255) / 256, 256, 0, stream>>>(
            fW1, fW2, cW1, cW2,
            fb1, fg1, fbe1, fm1, fv1, fb2, fg2, fbe2, fm2, fv2,
            cb1, cg1, cbe1, cm1, cv1, cb2, cg2, cbe2, cm2, cv2,
            lb, w1t, w2t, cw1t, cw2t, ss, outAcc, G);
    }

    const int mlpBlocks = (N + 127) / 128;
    const int roBlocks  = (N + 7) / 8;

    // layer 0: first_h MLP + readout (lb per node)
    mlp_kernel<true><<<mlpBlocks, 256, 0, stream>>>(
        x, nullptr, nullptr, 0, w1t, w2t, ss, ss + 256, hA, N);
    readout_kernel<<<roBlocks, 256, 0, stream>>>(hA, batch, lW, lb, 0, 1, outAcc, N);

    float* cur = hA;
    float* nxt = hB;
    for (int l = 0; l < 4; ++l) {
        hipMemsetAsync(aggr, 0, (size_t)N * HDIM * 4, stream);
        aggregate_kernel<<<(E + 7) / 8, 256, 0, stream>>>(cur, ei, aggr, E);
        mlp_kernel<false><<<mlpBlocks, 256, 0, stream>>>(
            cur, aggr, eps, l,
            cw1t + (size_t)l * HDIM * HDIM, cw2t + (size_t)l * HDIM * HDIM,
            ss + (2 + 2 * l) * 256, ss + (3 + 2 * l) * 256, nxt, N);
        readout_kernel<<<roBlocks, 256, 0, stream>>>(nxt, batch, lW, lb, l + 1, 0, outAcc, N);
        float* t = cur; cur = nxt; nxt = t;
    }

    sigmoid_kernel<<<(G + 255) / 256, 256, 0, stream>>>(outAcc, (float*)d_out, G);
}
